// RN_22162031247864
// MI455X (gfx1250) — matrix operand order repaired and checked by reading, not compiled, not run on hardware
//
#include <hip/hip_runtime.h>
#include <hip/hip_bf16.h>

// ---------------------------------------------------------------------------
// Relation Network forward for MI455X (gfx1250): all GEMMs via
// v_wmma_f32_16x16x32_f16 (wave32; 16x16x32 f16 tiles, f32 accumulate).
// Each wave owns a 32(M) x 16*NT(N) strip: 2 A fragments x NT B fragments,
// fully unrolled compile-time K loop -> straight-line load/wmma stream.
// ---------------------------------------------------------------------------

typedef __attribute__((ext_vector_type(16))) _Float16 v16h;
typedef __attribute__((ext_vector_type(8)))  _Float16 v8h;
typedef __attribute__((ext_vector_type(8)))  float    v8f;

#define RN_B      256
#define RN_NF     20
#define RN_T      16
#define RN_E      128
#define RN_H      256
#define RN_4H     1024
#define RN_KPAD   288      // 276 (H + 20 one-hot) padded to multiple of 32
#define RN_PAIRS  (RN_NF * RN_NF)              // 400
#define RN_PROWS  (RN_B * RN_PAIRS)            // 102400
#define RN_FROWS  (RN_B * RN_NF)               // 5120

// ---- WMMA fragment loads (gfx1250 16-bit layouts, CDNA5 ISA) ----

// A tile 16x32 (MxK), row-major source, lane L = row (L&15),
// K chunks at (L>>4)*8 and 16+(L>>4)*8.
__device__ __forceinline__ v16h load_a_frag(const _Float16* __restrict__ A,
                                            size_t lda, int k, int lane) {
  const _Float16* p = A + (size_t)(lane & 15) * lda + k + ((lane >> 4) << 3);
  v8h lo = *reinterpret_cast<const v8h*>(p);
  v8h hi = *reinterpret_cast<const v8h*>(p + 16);
  return __builtin_shufflevector(lo, hi, 0, 1, 2, 3, 4, 5, 6, 7,
                                 8, 9, 10, 11, 12, 13, 14, 15);
}

// B tile 32x16 (KxN) where B = W.T and W is [N,K] row-major: column n of B is
// row n of W -> K chunks at (L>>4)*8 and 16+(L>>4)*8 (same order as A).
__device__ __forceinline__ v16h load_bT_frag(const _Float16* __restrict__ W,
                                             size_t ldw, int n0, int k, int lane) {
  const _Float16* p = W + (size_t)(n0 + (lane & 15)) * ldw + k + ((lane >> 4) << 3);
  v8h lo = *reinterpret_cast<const v8h*>(p);
  v8h hi = *reinterpret_cast<const v8h*>(p + 16);
  return __builtin_shufflevector(lo, hi, 0, 1, 2, 3, 4, 5, 6, 7,
                                 8, 9, 10, 11, 12, 13, 14, 15);
}

// ---- generic GEMM:  out = act( A0 @ W0.T [+ A1 @ W1.T] + bias ) ----
// W row-major [N, K]. Wave tile: 32(M) x 16*NT(N). Block: 4 waves stacked in M
// -> 128 x 16*NT per block. Grids are exact (no bounds checks anywhere).
template <int NT, int K0, int K1, bool DUAL, bool RELU, bool OUT16>
__global__ __launch_bounds__(128) void gemm_wmma(
    const _Float16* __restrict__ A0, int lda0,
    const _Float16* __restrict__ W0,
    const _Float16* __restrict__ A1, int lda1,
    const _Float16* __restrict__ W1,
    const float* __restrict__ bias,
    void* __restrict__ out, int ldc) {
  const int lane = threadIdx.x & 31;
  const int wave = threadIdx.x >> 5;
  const int m0 = (blockIdx.y * 4 + wave) * 32;
  const int n0 = blockIdx.x * (16 * NT);

  v8f acc[2][NT];
#pragma unroll
  for (int i = 0; i < 2; ++i)
#pragma unroll
    for (int j = 0; j < NT; ++j)
#pragma unroll
      for (int r = 0; r < 8; ++r) acc[i][j][r] = 0.0f;

  {
    const _Float16* Ab = A0 + (size_t)m0 * lda0;
#pragma unroll
    for (int k = 0; k < K0; k += 32) {
      v16h a[2], b[NT];
#pragma unroll
      for (int i = 0; i < 2; ++i)
        a[i] = load_a_frag(Ab + (size_t)(i * 16) * lda0, lda0, k, lane);
#pragma unroll
      for (int j = 0; j < NT; ++j)
        b[j] = load_bT_frag(W0, K0, n0 + j * 16, k, lane);
#pragma unroll
      for (int i = 0; i < 2; ++i)
#pragma unroll
        for (int j = 0; j < NT; ++j)
          acc[i][j] = __builtin_amdgcn_wmma_f32_16x16x32_f16(
              false, a[i], false, b[j], (short)0, acc[i][j], false, false);
    }
  }
  if constexpr (DUAL) {
    const _Float16* Ab = A1 + (size_t)m0 * lda1;
#pragma unroll
    for (int k = 0; k < K1; k += 32) {
      v16h a[2], b[NT];
#pragma unroll
      for (int i = 0; i < 2; ++i)
        a[i] = load_a_frag(Ab + (size_t)(i * 16) * lda1, lda1, k, lane);
#pragma unroll
      for (int j = 0; j < NT; ++j)
        b[j] = load_bT_frag(W1, K1, n0 + j * 16, k, lane);
#pragma unroll
      for (int i = 0; i < 2; ++i)
#pragma unroll
        for (int j = 0; j < NT; ++j)
          acc[i][j] = __builtin_amdgcn_wmma_f32_16x16x32_f16(
              false, a[i], false, b[j], (short)0, acc[i][j], false, false);
    }
  }

  // C/D layout: lane L, VGPR r -> M = r + 8*(L>=16), N = L&15.
  const int ncol = lane & 15;
  const int mbase = (lane >> 4) << 3;
#pragma unroll
  for (int i = 0; i < 2; ++i) {
#pragma unroll
    for (int j = 0; j < NT; ++j) {
      const int n = n0 + j * 16 + ncol;
      const float bv = bias ? bias[n] : 0.0f;
      const size_t rowb = (size_t)(m0 + i * 16 + mbase);
#pragma unroll
      for (int r = 0; r < 8; ++r) {
        float v = acc[i][j][r] + bv;
        if (RELU) v = fmaxf(v, 0.0f);
        if (OUT16)
          ((_Float16*)out)[(rowb + r) * ldc + n] = (_Float16)v;
        else
          ((float*)out)[(rowb + r) * ldc + n] = v;
      }
    }
  }
}

// ---- elementwise / prep kernels ----

__global__ void cvt_f32_f16(const float* __restrict__ in,
                            _Float16* __restrict__ out, int n) {
  int i = blockIdx.x * blockDim.x + threadIdx.x;
  if (i < n) out[i] = (_Float16)in[i];
}

__global__ void add_bias_pair(const float* __restrict__ a,
                              const float* __restrict__ b,
                              float* __restrict__ out, int n) {
  int i = blockIdx.x * blockDim.x + threadIdx.x;
  if (i < n) out[i] = a[i] + b[i];
}

__global__ void fill_zero_f16(_Float16* __restrict__ p, int n) {
  int i = blockIdx.x * blockDim.x + threadIdx.x;
  if (i < n) p[i] = (_Float16)0.0f;
}

__global__ void fill_zero_f32(float* __restrict__ p, int n) {
  int i = blockIdx.x * blockDim.x + threadIdx.x;
  if (i < n) p[i] = 0.0f;
}

__global__ void gather_embed(const int* __restrict__ tok,
                             const float* __restrict__ emb,
                             _Float16* __restrict__ out, int ntok) {
  int i = blockIdx.x * blockDim.x + threadIdx.x;
  if (i >= ntok * RN_E) return;
  int t = i >> 7, e = i & (RN_E - 1);
  out[i] = (_Float16)emb[(size_t)tok[t] * RN_E + e];
}

__device__ __forceinline__ float sigf(float x) { return 1.0f / (1.0f + expf(-x)); }

// gates [M,4H] f32 (bias already added), PyTorch gate order i,f,g,o.
__global__ void lstm_cell(const float* __restrict__ gates,
                          float* __restrict__ c, _Float16* __restrict__ h16,
                          int M) {
  int i = blockIdx.x * blockDim.x + threadIdx.x;
  if (i >= M * RN_H) return;
  int m = i >> 8, j = i & (RN_H - 1);
  const float* g = gates + (size_t)m * RN_4H;
  float gi = sigf(g[j]);
  float gf = sigf(g[RN_H + j]);
  float gg = tanhf(g[2 * RN_H + j]);
  float go = sigf(g[3 * RN_H + j]);
  float cn = gf * c[i] + gi * gg;
  c[i] = cn;
  h16[i] = (_Float16)(go * tanhf(cn));
}

// split g_W0 [256,808] into W0a/W0b [256,288] (zero-padded from 276) + W0c [256,256]
__global__ void prep_w0(const float* __restrict__ gW0,
                        _Float16* __restrict__ w0a, _Float16* __restrict__ w0b,
                        _Float16* __restrict__ w0c) {
  int i = blockIdx.x * blockDim.x + threadIdx.x;
  if (i >= 256 * RN_KPAD) return;
  int n = i / RN_KPAD, k = i % RN_KPAD;
  const float* row = gW0 + (size_t)n * 808;
  w0a[i] = (k < 276) ? (_Float16)row[k] : (_Float16)0.0f;
  w0b[i] = (k < 276) ? (_Float16)row[276 + k] : (_Float16)0.0f;
  if (k < 256) w0c[n * 256 + k] = (_Float16)row[552 + k];
}

// fhcat [5120, 288]: [ h_f | one-hot(fact idx, 20) | zero pad ]
__global__ void build_fhcat(const _Float16* __restrict__ hf,
                            _Float16* __restrict__ fhcat) {
  int i = blockIdx.x * blockDim.x + threadIdx.x;
  if (i >= RN_FROWS * RN_KPAD) return;
  int r = i / RN_KPAD, col = i % RN_KPAD;
  _Float16 v = (_Float16)0.0f;
  if (col < RN_H) v = hf[(size_t)r * RN_H + col];
  else if (col - RN_H == (r % RN_NF)) v = (_Float16)1.0f;
  fhcat[i] = v;
}

// g layer 0 (factored): out[b*400+p] = relu(U[b*20+i] + V[b*20+j] + Q[b] + b0),
// i = p % 20 (oi cycles fast), j = p / 20 (oj repeats).
__global__ void build_pairs(const float* __restrict__ U,
                            const float* __restrict__ V,
                            const float* __restrict__ Q,
                            const float* __restrict__ b0,
                            _Float16* __restrict__ out) {
  int idx = blockIdx.x * blockDim.x + threadIdx.x;
  if (idx >= RN_PROWS * 256) return;
  int row = idx >> 8, ch = idx & 255;
  int b = row / RN_PAIRS, p = row % RN_PAIRS;
  int i = p % RN_NF, j = p / RN_NF;
  float v = U[(size_t)(b * RN_NF + i) * 256 + ch] +
            V[(size_t)(b * RN_NF + j) * 256 + ch] +
            Q[(size_t)b * 256 + ch] + b0[ch];
  out[idx] = (_Float16)fmaxf(v, 0.0f);
}

// rel_sum[b, ch] = sum_p rel[b*400+p, ch], f32 accumulate -> f16 out
__global__ void reduce_pairs(const _Float16* __restrict__ rel,
                             _Float16* __restrict__ out16) {
  int b = blockIdx.x, ch = threadIdx.x;
  const _Float16* base = rel + (size_t)b * RN_PAIRS * 256 + ch;
  float s = 0.0f;
  for (int p = 0; p < RN_PAIRS; ++p) s += (float)base[(size_t)p * 256];
  out16[b * 256 + ch] = (_Float16)s;
}

// ---------------------------------------------------------------------------

static inline void* wscarve(void* base, size_t& off, size_t bytes) {
  void* p = (char*)base + off;
  off += (bytes + 255) & ~(size_t)255;
  return p;
}
static inline int cdiv(int a, int b) { return (a + b - 1) / b; }

extern "C" void kernel_launch(void* const* d_in, const int* in_sizes, int n_in,
                              void* d_out, int out_size, void* d_ws, size_t ws_size,
                              hipStream_t stream) {
  (void)in_sizes; (void)n_in; (void)out_size; (void)ws_size;

  const int*   q_tok = (const int*)d_in[0];
  const int*   f_tok = (const int*)d_in[1];
  const float* emb   = (const float*)d_in[2];
  const float* Wih_q = (const float*)d_in[3];
  const float* Whh_q = (const float*)d_in[4];
  const float* bih_q = (const float*)d_in[5];
  const float* bhh_q = (const float*)d_in[6];
  const float* Wih_f = (const float*)d_in[7];
  const float* Whh_f = (const float*)d_in[8];
  const float* bih_f = (const float*)d_in[9];
  const float* bhh_f = (const float*)d_in[10];
  const float* gW0 = (const float*)d_in[11]; const float* gb0 = (const float*)d_in[12];
  const float* gW1 = (const float*)d_in[13]; const float* gb1 = (const float*)d_in[14];
  const float* gW2 = (const float*)d_in[15]; const float* gb2 = (const float*)d_in[16];
  const float* gW3 = (const float*)d_in[17]; const float* gb3 = (const float*)d_in[18];
  const float* fW0 = (const float*)d_in[19]; const float* fb0 = (const float*)d_in[20];
  const float* fW1 = (const float*)d_in[21]; const float* fb1 = (const float*)d_in[22];
  const float* fW2 = (const float*)d_in[23]; const float* fb2 = (const float*)d_in[24];

  // ---- workspace carving ----
  size_t off = 0;
  _Float16* xq     = (_Float16*)wscarve(d_ws, off, (size_t)RN_B * RN_T * RN_E * 2);
  _Float16* xf     = (_Float16*)wscarve(d_ws, off, (size_t)RN_FROWS * RN_T * RN_E * 2);
  _Float16* wihq16 = (_Float16*)wscarve(d_ws, off, (size_t)RN_4H * RN_E * 2);
  _Float16* whhq16 = (_Float16*)wscarve(d_ws, off, (size_t)RN_4H * RN_H * 2);
  _Float16* wihf16 = (_Float16*)wscarve(d_ws, off, (size_t)RN_4H * RN_E * 2);
  _Float16* whhf16 = (_Float16*)wscarve(d_ws, off, (size_t)RN_4H * RN_H * 2);
  float*    bsq    = (float*)wscarve(d_ws, off, (size_t)RN_4H * 4);
  float*    bsf    = (float*)wscarve(d_ws, off, (size_t)RN_4H * 4);
  float*    gates  = (float*)wscarve(d_ws, off, (size_t)RN_FROWS * RN_4H * 4);
  _Float16* hq     = (_Float16*)wscarve(d_ws, off, (size_t)RN_B * RN_H * 2);
  float*    cq     = (float*)wscarve(d_ws, off, (size_t)RN_B * RN_H * 4);
  _Float16* hf     = (_Float16*)wscarve(d_ws, off, (size_t)RN_FROWS * RN_H * 2);
  float*    cf     = (float*)wscarve(d_ws, off, (size_t)RN_FROWS * RN_H * 4);
  _Float16* fhcat  = (_Float16*)wscarve(d_ws, off, (size_t)RN_FROWS * RN_KPAD * 2);
  _Float16* w0a    = (_Float16*)wscarve(d_ws, off, (size_t)256 * RN_KPAD * 2);
  _Float16* w0b    = (_Float16*)wscarve(d_ws, off, (size_t)256 * RN_KPAD * 2);
  _Float16* w0c    = (_Float16*)wscarve(d_ws, off, (size_t)256 * 256 * 2);
  _Float16* gw1_16 = (_Float16*)wscarve(d_ws, off, (size_t)256 * 256 * 2);
  _Float16* gw2_16 = (_Float16*)wscarve(d_ws, off, (size_t)256 * 256 * 2);
  _Float16* gw3_16 = (_Float16*)wscarve(d_ws, off, (size_t)256 * 256 * 2);
  _Float16* fw0_16 = (_Float16*)wscarve(d_ws, off, (size_t)256 * 256 * 2);
  _Float16* fw1_16 = (_Float16*)wscarve(d_ws, off, (size_t)512 * 256 * 2);
  _Float16* fw2_16 = (_Float16*)wscarve(d_ws, off, (size_t)32 * 512 * 2);
  float*    U      = (float*)wscarve(d_ws, off, (size_t)RN_FROWS * 256 * 4);
  float*    V      = (float*)wscarve(d_ws, off, (size_t)RN_FROWS * 256 * 4);
  float*    Qp     = (float*)wscarve(d_ws, off, (size_t)RN_B * 256 * 4);
  _Float16* pA     = (_Float16*)wscarve(d_ws, off, (size_t)RN_PROWS * 256 * 2);
  _Float16* pB     = (_Float16*)wscarve(d_ws, off, (size_t)RN_PROWS * 256 * 2);
  _Float16* relsum = (_Float16*)wscarve(d_ws, off, (size_t)RN_B * 256 * 2);
  _Float16* fx1    = (_Float16*)wscarve(d_ws, off, (size_t)RN_B * 256 * 2);
  _Float16* fx2    = (_Float16*)wscarve(d_ws, off, (size_t)RN_B * 512 * 2);

  const int TB = 256;

  // ---- weight prep (f32 -> f16) ----
  cvt_f32_f16<<<cdiv(RN_4H * RN_E, TB), TB, 0, stream>>>(Wih_q, wihq16, RN_4H * RN_E);
  cvt_f32_f16<<<cdiv(RN_4H * RN_H, TB), TB, 0, stream>>>(Whh_q, whhq16, RN_4H * RN_H);
  cvt_f32_f16<<<cdiv(RN_4H * RN_E, TB), TB, 0, stream>>>(Wih_f, wihf16, RN_4H * RN_E);
  cvt_f32_f16<<<cdiv(RN_4H * RN_H, TB), TB, 0, stream>>>(Whh_f, whhf16, RN_4H * RN_H);
  cvt_f32_f16<<<cdiv(256 * 256, TB), TB, 0, stream>>>(gW1, gw1_16, 256 * 256);
  cvt_f32_f16<<<cdiv(256 * 256, TB), TB, 0, stream>>>(gW2, gw2_16, 256 * 256);
  cvt_f32_f16<<<cdiv(256 * 256, TB), TB, 0, stream>>>(gW3, gw3_16, 256 * 256);
  cvt_f32_f16<<<cdiv(256 * 256, TB), TB, 0, stream>>>(fW0, fw0_16, 256 * 256);
  cvt_f32_f16<<<cdiv(512 * 256, TB), TB, 0, stream>>>(fW1, fw1_16, 512 * 256);
  cvt_f32_f16<<<cdiv(32 * 512, TB), TB, 0, stream>>>(fW2, fw2_16, 32 * 512);
  add_bias_pair<<<cdiv(RN_4H, TB), TB, 0, stream>>>(bih_q, bhh_q, bsq, RN_4H);
  add_bias_pair<<<cdiv(RN_4H, TB), TB, 0, stream>>>(bih_f, bhh_f, bsf, RN_4H);
  prep_w0<<<cdiv(256 * RN_KPAD, TB), TB, 0, stream>>>(gW0, w0a, w0b, w0c);

  // ---- embedding gather (-> f16) ----
  gather_embed<<<cdiv(RN_B * RN_T * RN_E, TB), TB, 0, stream>>>(q_tok, emb, xq, RN_B * RN_T);
  gather_embed<<<cdiv(RN_FROWS * RN_T * RN_E, TB), TB, 0, stream>>>(f_tok, emb, xf, RN_FROWS * RN_T);

  // ---- LSTM init ----
  fill_zero_f16<<<cdiv(RN_B * RN_H, TB), TB, 0, stream>>>(hq, RN_B * RN_H);
  fill_zero_f32<<<cdiv(RN_B * RN_H, TB), TB, 0, stream>>>(cq, RN_B * RN_H);
  fill_zero_f16<<<cdiv(RN_FROWS * RN_H, TB), TB, 0, stream>>>(hf, RN_FROWS * RN_H);
  fill_zero_f32<<<cdiv(RN_FROWS * RN_H, TB), TB, 0, stream>>>(cf, RN_FROWS * RN_H);

  // ---- LSTM steps: gates = x_t@Wih.T + h@Whh.T + (bih+bhh), then cell ----
  for (int t = 0; t < RN_T; ++t) {
    gemm_wmma<4, RN_E, RN_H, true, false, false>
        <<<dim3(RN_4H / 64, RN_FROWS / 128), 128, 0, stream>>>(
            xf + (size_t)t * RN_E, RN_T * RN_E, wihf16,
            hf, RN_H, whhf16, bsf, gates, RN_4H);
    lstm_cell<<<cdiv(RN_FROWS * RN_H, TB), TB, 0, stream>>>(gates, cf, hf, RN_FROWS);

    gemm_wmma<4, RN_E, RN_H, true, false, false>
        <<<dim3(RN_4H / 64, RN_B / 128), 128, 0, stream>>>(
            xq + (size_t)t * RN_E, RN_T * RN_E, wihq16,
            hq, RN_H, whhq16, bsq, gates, RN_4H);
    lstm_cell<<<cdiv(RN_B * RN_H, TB), TB, 0, stream>>>(gates, cq, hq, RN_B);
  }

  // ---- g layer 0, factored: U = fhcat@W0a.T, V = fhcat@W0b.T, Qp = hq@W0c.T
  build_fhcat<<<cdiv(RN_FROWS * RN_KPAD, TB), TB, 0, stream>>>(hf, fhcat);
  gemm_wmma<4, RN_KPAD, 32, false, false, false>
      <<<dim3(256 / 64, RN_FROWS / 128), 128, 0, stream>>>(
          fhcat, RN_KPAD, w0a, nullptr, 0, nullptr, nullptr, U, 256);
  gemm_wmma<4, RN_KPAD, 32, false, false, false>
      <<<dim3(256 / 64, RN_FROWS / 128), 128, 0, stream>>>(
          fhcat, RN_KPAD, w0b, nullptr, 0, nullptr, nullptr, V, 256);
  gemm_wmma<4, RN_H, 32, false, false, false>
      <<<dim3(256 / 64, RN_B / 128), 128, 0, stream>>>(
          hq, RN_H, w0c, nullptr, 0, nullptr, nullptr, Qp, 256);
  build_pairs<<<cdiv(RN_PROWS * 256, TB), TB, 0, stream>>>(U, V, Qp, gb0, pA);

  // ---- g layers 1..3: 102400 x 256 x 256 WMMA GEMMs with fused bias+relu ----
  gemm_wmma<4, 256, 32, false, true, true>
      <<<dim3(256 / 64, RN_PROWS / 128), 128, 0, stream>>>(
          pA, 256, gw1_16, nullptr, 0, nullptr, gb1, pB, 256);
  gemm_wmma<4, 256, 32, false, true, true>
      <<<dim3(256 / 64, RN_PROWS / 128), 128, 0, stream>>>(
          pB, 256, gw2_16, nullptr, 0, nullptr, gb2, pA, 256);
  gemm_wmma<4, 256, 32, false, true, true>
      <<<dim3(256 / 64, RN_PROWS / 128), 128, 0, stream>>>(
          pA, 256, gw3_16, nullptr, 0, nullptr, gb3, pB, 256);

  // ---- sum over the 400 pairs ----
  reduce_pairs<<<RN_B, 256, 0, stream>>>(pB, relsum);

  // ---- f MLP: 256 -> 256 -> 512 -> 32, relu everywhere, final out f32 ----
  gemm_wmma<4, 256, 32, false, true, true>
      <<<dim3(256 / 64, RN_B / 128), 128, 0, stream>>>(
          relsum, 256, fw0_16, nullptr, 0, nullptr, fb0, fx1, 256);
  gemm_wmma<4, 256, 32, false, true, true>
      <<<dim3(512 / 64, RN_B / 128), 128, 0, stream>>>(
          fx1, 256, fw1_16, nullptr, 0, nullptr, fb1, fx2, 512);
  gemm_wmma<2, 512, 32, false, true, false>
      <<<dim3(1, RN_B / 128), 128, 0, stream>>>(
          fx2, 512, fw2_16, nullptr, 0, nullptr, fb2, d_out, 32);
}
